// generator_80436147520070
// MI455X (gfx1250) — matrix operand order repaired and checked by reading, not compiled, not run on hardware
//
#include <hip/hip_runtime.h>
#include <hip/hip_bf16.h>
#include <stdint.h>

// ---------------------------------------------------------------------------
// CDNA5 (gfx1250) wave32 WMMA bf16 pipeline for hetero-SAGE + diffusion + MLP
// A-tile staging uses the CDNA5 async global->LDS path (ASYNCcnt).
// ---------------------------------------------------------------------------

typedef __attribute__((ext_vector_type(16))) __bf16 bf16x16;
typedef __attribute__((ext_vector_type(8)))  float  f32x8;

#define EP_NONE 0
#define EP_RELU 1
#define EP_SIGMOID 2

__device__ __forceinline__ unsigned short f2bf_rne(float f) {
  unsigned int u = __float_as_uint(f);
  unsigned int r = u + 0x7FFFu + ((u >> 16) & 1u);   // round-to-nearest-even
  return (unsigned short)(r >> 16);
}

// ---------------------------------------------------------------------------
// GEMM: C[M,N] (f32) = epilogue( A[M,K](bf16) @ B[K,N](bf16) + bias + addC*C )
// lda=K, ldb=N, ldc=N. Requires K%32==0, N%64==0; M guarded (clamped loads,
// guarded stores). Block tile 64x64, BK=32, 256 threads = 8 waves; each wave
// computes a 16x32 slab of C (two v_wmma_f32_16x16x32_bf16 per K step).
// ---------------------------------------------------------------------------
__global__ void __launch_bounds__(256) gemm_bf16_kernel(
    const unsigned short* __restrict__ A,
    const unsigned short* __restrict__ B,
    const float* __restrict__ bias,
    float* __restrict__ C,
    int M, int N, int K, int epilogue, int addC)
{
  // As: async-filled; rows are 64B so each lane's 16B target is 16B-aligned.
  __shared__ __align__(16) unsigned short As[64][32];   // [m][k]
  __shared__ unsigned short Bt[64][34];                 // transposed: [n][k]

  const int tid  = threadIdx.x;
  const int lane = tid & 31;
  const int wid  = tid >> 5;
  const int wrow = wid >> 1;        // 0..3 -> 16-row slab
  const int wcol = wid & 1;         // 0..1 -> 32-col slab
  const int half = lane >> 4;       // lane half per ISA 7.12.2 layouts
  const int l16  = lane & 15;
  const int blockM = blockIdx.y * 64;
  const int blockN = blockIdx.x * 64;

  f32x8 acc0 = {};
  f32x8 acc1 = {};

  // A-tile loader: 64 rows x 4 groups of 8 bf16 (16 bytes per lane).
  const int ar = tid >> 2, ac = (tid & 3) * 8;
  int gmA = blockM + ar;
  if (gmA >= M) gmA = M - 1;  // clamp: rows >= M never stored, data harmless
  const unsigned short* gA = A + (size_t)gmA * K + ac;
  // Low 32 bits of the LDS-aperture generic address = workgroup LDS offset.
  const unsigned ldsA = (unsigned)(uintptr_t)(&As[ar][ac]);

  // B-tile loader: 32 k-rows x 8 groups of 8 bf16.
  const int br = tid >> 3, bc = (tid & 7) * 8;

  for (int k0 = 0; k0 < K; k0 += 32) {
    // ---- stage A 64x32 via async global->LDS DMA (ASYNCcnt) ----
    asm volatile("global_load_async_to_lds_b128 %0, %1, off"
                 :
                 : "v"(ldsA), "v"((unsigned long long)(uintptr_t)(gA + k0))
                 : "memory");
    // ---- stage B 32x64, transposed into LDS ----
    {
      uint4 v = *(const uint4*)(B + (size_t)(k0 + br) * N + blockN + bc);
      unsigned int u[4] = {v.x, v.y, v.z, v.w};
      #pragma unroll
      for (int q = 0; q < 4; ++q) {
        Bt[bc + 2 * q + 0][br] = (unsigned short)(u[q] & 0xFFFFu);
        Bt[bc + 2 * q + 1][br] = (unsigned short)(u[q] >> 16);
      }
    }
    if (k0 + 32 < K)
      __builtin_prefetch(B + (size_t)(k0 + 32 + br) * N + blockN + bc, 0, 1);
    // All of this wave's async LDS writes must land before the barrier.
    asm volatile("s_wait_asynccnt 0x0" ::: "memory");
    __syncthreads();

    // ---- A fragment (16x32 bf16, ISA 7.12.2 K split across lane halves) ----
    union { unsigned int u[8]; bf16x16 v; } af;
    {
      int m = wrow * 16 + l16;
      #pragma unroll
      for (int j = 0; j < 4; ++j)
        af.u[j] = *(const unsigned int*)&As[m][half * 8 + 2 * j];
      #pragma unroll
      for (int j = 0; j < 4; ++j)
        af.u[4 + j] = *(const unsigned int*)&As[m][16 + half * 8 + 2 * j];
    }
    // ---- B fragments for the wave's two 16-col tiles + WMMA ----
    union { unsigned int u[8]; bf16x16 v; } bfrag;
    {
      int n = wcol * 32 + l16;
      #pragma unroll
      for (int j = 0; j < 8; ++j)
        bfrag.u[j] = *(const unsigned int*)&Bt[n][(j >> 2) * 16 + half * 8 + 2 * (j & 3)];
      acc0 = __builtin_amdgcn_wmma_f32_16x16x32_bf16(
          false, af.v, false, bfrag.v, (short)0, acc0, false, false);
      n += 16;
      #pragma unroll
      for (int j = 0; j < 8; ++j)
        bfrag.u[j] = *(const unsigned int*)&Bt[n][(j >> 2) * 16 + half * 8 + 2 * (j & 3)];
      acc1 = __builtin_amdgcn_wmma_f32_16x16x32_bf16(
          false, af.v, false, bfrag.v, (short)0, acc1, false, false);
    }
    __syncthreads();
  }

  // ---- epilogue: C/D layout VGPR r -> M = r + 8*half, N = lane&15 ----
  #pragma unroll
  for (int tile = 0; tile < 2; ++tile) {
    f32x8 acc = tile ? acc1 : acc0;
    int n = blockN + wcol * 32 + tile * 16 + l16;
    float bv = bias ? bias[n] : 0.0f;
    #pragma unroll
    for (int r = 0; r < 8; ++r) {
      int m = blockM + wrow * 16 + half * 8 + r;
      if (m < M) {
        size_t idx = (size_t)m * N + n;
        float v = acc[r] + bv;
        if (addC) v += C[idx];
        if (epilogue == EP_RELU)         v = fmaxf(v, 0.0f);
        else if (epilogue == EP_SIGMOID) v = 1.0f / (1.0f + __expf(-v));
        C[idx] = v;
      }
    }
  }
}

// ---------------------------------------------------------------------------
// Edge segment-max scatter. pooled is post-ReLU (>= 0) so IEEE bit patterns
// are monotone as u32 -> atomicMax(u32) == float max. agg pre-zeroed (matches
// DGL's 0 for isolated destinations).
// ---------------------------------------------------------------------------
__global__ void __launch_bounds__(256) scatter_max_kernel(
    const float* __restrict__ pooled, const int* __restrict__ src,
    const int* __restrict__ dst, float* __restrict__ agg, int E, int F)
{
  const int chunks = F >> 2;
  long long idx = (long long)blockIdx.x * blockDim.x + threadIdx.x;
  if (idx >= (long long)E * chunks) return;
  int e = (int)(idx / chunks);
  int c = (int)(idx % chunks);
  int s = src[e], d = dst[e];
  const float4 v = *(const float4*)(pooled + (size_t)s * F + c * 4);
  unsigned int* a = (unsigned int*)(agg + (size_t)d * F + c * 4);
  atomicMax(a + 0, __float_as_uint(v.x));
  atomicMax(a + 1, __float_as_uint(v.y));
  atomicMax(a + 2, __float_as_uint(v.z));
  atomicMax(a + 3, __float_as_uint(v.w));
}

__global__ void f32_to_bf16_flat(const float* __restrict__ s,
                                 unsigned short* __restrict__ d, long long n)
{
  long long i = (long long)blockIdx.x * blockDim.x + threadIdx.x;
  if (i < n) d[i] = f2bf_rne(s[i]);
}

__global__ void f32_to_bf16_pitched(const float* __restrict__ s, int sld,
                                    unsigned short* __restrict__ d, int dld,
                                    long long rows, int cols)
{
  long long i = (long long)blockIdx.x * blockDim.x + threadIdx.x;
  if (i >= rows * (long long)cols) return;
  long long r = i / cols;
  int c = (int)(i % cols);
  d[r * (long long)dld + c] = f2bf_rne(s[r * (long long)sld + c]);
}

// ---------------------------------------------------------------------------
// Fused L1-normalize + diffusion-noise + slice. One wave per output row
// (OUT=64, each lane owns 2 features). Writes f32 fake to d_out and bf16
// fake columns into the MLP concat buffer. t / leftIndex read on-device.
// ---------------------------------------------------------------------------
__global__ void __launch_bounds__(256) fake_kernel(
    const float* __restrict__ mirna, const float* __restrict__ noise,
    const int* __restrict__ t_p, const int* __restrict__ left_p,
    float* __restrict__ fake_out, unsigned short* __restrict__ mcat,
    int mcat_ld, int fake_col0, int n_rows)
{
  int lane = threadIdx.x & 31;
  int wr = (int)(((long long)blockIdx.x * blockDim.x + threadIdx.x) >> 5);
  if (wr >= n_rows) return;
  int row = *left_p + wr;
  float a = mirna[(size_t)row * 64 + lane];
  float b = mirna[(size_t)row * 64 + 32 + lane];
  float s = fabsf(a) + fabsf(b);
  #pragma unroll
  for (int off = 16; off; off >>= 1) s += __shfl_xor(s, off, 32);
  float denom = fmaxf(s, 1e-12f);
  int t = *t_p;
  float ab = 1.0f;
  for (int i = 0; i <= t; ++i) {                 // cumprod(1-beta), T=100
    float beta = 1e-4f + (0.02f - 1e-4f) * ((float)i / 99.0f);
    ab *= (1.0f - beta);
  }
  float sr = sqrtf(ab), nr = sqrtf(1.0f - ab);
  float f0 = sr * (a / denom) + nr * noise[(size_t)row * 64 + lane];
  float f1 = sr * (b / denom) + nr * noise[(size_t)row * 64 + 32 + lane];
  fake_out[(size_t)wr * 64 + lane] = f0;
  fake_out[(size_t)wr * 64 + 32 + lane] = f1;
  mcat[(size_t)wr * mcat_ld + fake_col0 + lane] = f2bf_rne(f0);
  mcat[(size_t)wr * mcat_ld + fake_col0 + 32 + lane] = f2bf_rne(f1);
}

// ---------------------------------------------------------------------------
// Host driver
// ---------------------------------------------------------------------------
extern "C" void kernel_launch(void* const* d_in, const int* in_sizes, int n_in,
                              void* d_out, int out_size, void* d_ws, size_t ws_size,
                              hipStream_t stream) {
  (void)n_in; (void)out_size; (void)ws_size;
  static const int SRCr[6] = {0, 1, 0, 2, 2, 1};
  static const int DSTr[6] = {1, 0, 2, 0, 1, 2};
  const int F_IN = 128, HID = 64, SIZE = 4096, ITEMS = 4096;

  const float* hM = (const float*)d_in[0];
  const float* hD = (const float*)d_in[1];
  const float* hL = (const float*)d_in[2];
  const float* Wpool[3] = {(const float*)d_in[3], (const float*)d_in[8],  (const float*)d_in[13]};
  const float* bpool[3] = {(const float*)d_in[4], (const float*)d_in[9],  (const float*)d_in[14]};
  const float* Wself[3] = {(const float*)d_in[5], (const float*)d_in[10], (const float*)d_in[15]};
  const float* bself[3] = {(const float*)d_in[6], (const float*)d_in[11], (const float*)d_in[16]};
  const float* Wneig[3] = {(const float*)d_in[7], (const float*)d_in[12], (const float*)d_in[17]};
  const float* Adj   = (const float*)d_in[18];
  const float* noise = (const float*)d_in[19];
  const float* W1 = (const float*)d_in[20]; const float* b1 = (const float*)d_in[21];
  const float* W2 = (const float*)d_in[22]; const float* b2 = (const float*)d_in[23];
  const float* W3 = (const float*)d_in[24]; const float* b3 = (const float*)d_in[25];
  const float* W4 = (const float*)d_in[26]; const float* b4 = (const float*)d_in[27];
  const int* edges  = (const int*)d_in[28];
  const int* t_p    = (const int*)d_in[29];
  const int* left_p = (const int*)d_in[31];

  const int N = in_sizes[0] / F_IN;        // 50000
  const int E = in_sizes[28] / 12;         // 500000
  const int CATW = ITEMS + HID;            // 4160

  // ---- carve workspace ----
  size_t off = 0;
  auto carve = [&](size_t bytes) -> void* {
    void* p = (char*)d_ws + off;
    off += (bytes + 255) & ~(size_t)255;
    return p;
  };
  const int FinL[3]  = {F_IN, HID, HID};
  const int FoutL[3] = {HID, HID, HID};
  unsigned short* WpB[3]; unsigned short* WsB[3]; unsigned short* WnB[3];
  for (int L = 0; L < 3; ++L) {
    WpB[L] = (unsigned short*)carve((size_t)6 * FinL[L] * FinL[L] * 2);
    WsB[L] = (unsigned short*)carve((size_t)6 * FinL[L] * FoutL[L] * 2);
    WnB[L] = (unsigned short*)carve((size_t)6 * FinL[L] * FoutL[L] * 2);
  }
  unsigned short* W1b = (unsigned short*)carve((size_t)CATW * 256 * 2);
  unsigned short* W2b = (unsigned short*)carve((size_t)256 * 512 * 2);
  unsigned short* W3b = (unsigned short*)carve((size_t)512 * 1024 * 2);
  unsigned short* W4b = (unsigned short*)carve((size_t)1024 * ITEMS * 2);
  unsigned short* h_bf[3];
  for (int i = 0; i < 3; ++i) h_bf[i] = (unsigned short*)carve((size_t)N * F_IN * 2);
  float*          pooled    = (float*)carve((size_t)N * F_IN * 4);
  unsigned short* pooled_bf = (unsigned short*)carve((size_t)N * F_IN * 2);
  float*          agg       = (float*)carve((size_t)N * F_IN * 4);
  unsigned short* agg_bf    = (unsigned short*)carve((size_t)N * F_IN * 2);
  float* outP[2][3];
  for (int p = 0; p < 2; ++p)
    for (int i = 0; i < 3; ++i) outP[p][i] = (float*)carve((size_t)N * HID * 4);
  unsigned short* mcat = (unsigned short*)carve((size_t)SIZE * CATW * 2);
  float*          act1  = (float*)carve((size_t)SIZE * 256 * 4);
  unsigned short* act1b = (unsigned short*)carve((size_t)SIZE * 256 * 2);
  float*          act2  = (float*)carve((size_t)SIZE * 512 * 4);
  unsigned short* act2b = (unsigned short*)carve((size_t)SIZE * 512 * 2);
  float*          act3  = (float*)carve((size_t)SIZE * 1024 * 4);
  unsigned short* act3b = (unsigned short*)carve((size_t)SIZE * 1024 * 2);

  auto conv = [&](const float* s, unsigned short* d2, long long n) {
    f32_to_bf16_flat<<<(unsigned)((n + 255) / 256), 256, 0, stream>>>(s, d2, n);
  };
  auto gemm = [&](const unsigned short* A, const unsigned short* B, const float* bias,
                  float* C, int M, int Nn, int K, int ep, int addC) {
    dim3 g((unsigned)(Nn / 64), (unsigned)((M + 63) / 64));
    gemm_bf16_kernel<<<g, 256, 0, stream>>>(A, B, bias, C, M, Nn, K, ep, addC);
  };

  // ---- convert all weights to bf16 once per call ----
  for (int L = 0; L < 3; ++L) {
    conv(Wpool[L], WpB[L], (long long)6 * FinL[L] * FinL[L]);
    conv(Wself[L], WsB[L], (long long)6 * FinL[L] * FoutL[L]);
    conv(Wneig[L], WnB[L], (long long)6 * FinL[L] * FoutL[L]);
  }
  conv(W1, W1b, (long long)CATW * 256);
  conv(W2, W2b, (long long)256 * 512);
  conv(W3, W3b, (long long)512 * 1024);
  conv(W4, W4b, (long long)1024 * ITEMS);

  // ---- 3 hetero-SAGE layers ----
  const float* hin[3] = {hM, hD, hL};
  for (int L = 0; L < 3; ++L) {
    const int Fin = FinL[L], Fout = FoutL[L];
    for (int tt = 0; tt < 3; ++tt) conv(hin[tt], h_bf[tt], (long long)N * Fin);
    float** outb = outP[L & 1];
    for (int tt = 0; tt < 3; ++tt)
      hipMemsetAsync(outb[tt], 0, (size_t)N * Fout * 4, stream);
    for (int e = 0; e < 6; ++e) {
      const int s = SRCr[e], d = DSTr[e];
      // pooled = relu(h[s] @ Wp + bp)
      gemm(h_bf[s], WpB[L] + (size_t)e * Fin * Fin, bpool[L] + (size_t)e * Fin,
           pooled, N, Fin, Fin, EP_RELU, 0);
      conv(pooled, pooled_bf, (long long)N * Fin);
      // agg = segment_max over edges (exact fp32), isolated -> 0
      hipMemsetAsync(agg, 0, (size_t)N * Fin * 4, stream);
      {
        long long tot = (long long)E * (Fin / 4);
        scatter_max_kernel<<<(unsigned)((tot + 255) / 256), 256, 0, stream>>>(
            pooled, edges + (size_t)(2 * e) * E, edges + (size_t)(2 * e + 1) * E,
            agg, E, Fin);
      }
      conv(agg, agg_bf, (long long)N * Fin);
      // out[d] += h[d] @ Ws + bs ; out[d] += agg @ Wn
      gemm(h_bf[d], WsB[L] + (size_t)e * Fin * Fout, bself[L] + (size_t)e * Fout,
           outb[d], N, Fout, Fin, EP_NONE, 1);
      gemm(agg_bf, WnB[L] + (size_t)e * Fin * Fout, nullptr,
           outb[d], N, Fout, Fin, EP_NONE, 1);
    }
    hin[0] = outb[0]; hin[1] = outb[1]; hin[2] = outb[2];
  }

  // ---- fake = slice(noisy(L1norm(mirna))), fp32 to d_out + bf16 into concat ----
  float* fake_out = (float*)d_out;
  {
    long long thr = (long long)SIZE * 32;
    fake_kernel<<<(unsigned)((thr + 255) / 256), 256, 0, stream>>>(
        hin[0], noise, t_p, left_p, fake_out, mcat, CATW, ITEMS, SIZE);
  }
  // Adj -> bf16 concat columns [0, ITEMS)
  {
    long long n = (long long)SIZE * ITEMS;
    f32_to_bf16_pitched<<<(unsigned)((n + 255) / 256), 256, 0, stream>>>(
        Adj, ITEMS, mcat, CATW, SIZE, ITEMS);
  }

  // ---- MLP: relu x3 + sigmoid, final layer straight into d_out ----
  gemm(mcat,  W1b, b1, act1, SIZE, 256,  CATW, EP_RELU, 0);
  conv(act1, act1b, (long long)SIZE * 256);
  gemm(act1b, W2b, b2, act2, SIZE, 512,  256,  EP_RELU, 0);
  conv(act2, act2b, (long long)SIZE * 512);
  gemm(act2b, W3b, b3, act3, SIZE, 1024, 512,  EP_RELU, 0);
  conv(act3, act3b, (long long)SIZE * 1024);
  gemm(act3b, W4b, b4, (float*)d_out + (size_t)SIZE * HID,
       SIZE, ITEMS, 1024, EP_SIGMOID, 0);
}
